// MultiPoolModel_60619168416468
// MI455X (gfx1250) — compile-verified, matrix operand order checked by reading, not run on hardware
//
#include <hip/hip_runtime.h>

typedef __attribute__((ext_vector_type(16))) _Float16 v16h;
typedef __attribute__((ext_vector_type(8)))  _Float16 v8h;
typedef __attribute__((ext_vector_type(8)))  float    v8f;

#define HDIM 128
#define GNUM 256
#define CNUM 3

// ---- degree / normalization --------------------------------------------------
__global__ __launch_bounds__(256) void k_deg(const int* __restrict__ col,
                                             float* __restrict__ deg, int E) {
  int i = blockIdx.x * 256 + threadIdx.x;
  if (i < E) atomicAdd(&deg[col[i]], 1.0f);
}

__global__ __launch_bounds__(256) void k_dinv(float* __restrict__ deg, int N) {
  int i = blockIdx.x * 256 + threadIdx.x;
  if (i < N) deg[i] = rsqrtf(deg[i] + 1.0f);   // +1 self loop
}

// ---- f32 -> f16 (straight copy, for activations) -----------------------------
__global__ __launch_bounds__(256) void k_f32_to_f16(const float* __restrict__ in,
                                                    _Float16* __restrict__ out, int n) {
  int i = blockIdx.x * 256 + threadIdx.x;
  if (i < n) out[i] = (_Float16)in[i];
}

// ---- f32 [K,H] -> f16 transposed [H,K] (for WMMA B operand) ------------------
__global__ __launch_bounds__(256) void k_f32_to_f16_tr(const float* __restrict__ in,
                                                       _Float16* __restrict__ out) {
  int idx = blockIdx.x * 256 + threadIdx.x;          // idx over K*H
  if (idx >= HDIM * HDIM) return;
  int k = idx >> 7, n = idx & (HDIM - 1);
  out[n * HDIM + k] = (_Float16)in[idx];
}

// ---- GEMM: C[N,128] = A[N,128] x W[128,128]; Wt is column-major [H,K] --------
// One wave computes a 16x32 output tile (2 accumulators share the A fragment).
__global__ __launch_bounds__(256) void k_gemm_wmma(const _Float16* __restrict__ A,
                                                   const _Float16* __restrict__ Wt,
                                                   float* __restrict__ C, int N) {
  const int K = HDIM, H = HDIM;
  int wave = blockIdx.x * 8 + (threadIdx.x >> 5);
  int lane = threadIdx.x & 31;
  const int wavesPerRow = H / 32;                    // 4
  int mT = wave / wavesPerRow, nT = wave % wavesPerRow;
  int m0 = mT * 16, n0 = nT * 32;
  if (m0 >= N) return;
  int r  = lane & 15;                                // A row / B col within tile
  int hi = lane >> 4;                                // K sub-block select
  int am = m0 + r; if (am > N - 1) am = N - 1;       // clamp partial tile reads

  // Fragment K striping (16-bit A 16x32 layout, ISA 7.12.2):
  //   lanes 0-15 : K = {k0..k0+7, k0+16..k0+23}
  //   lanes 16-31: K = {k0+8..k0+15, k0+24..k0+31}
  const _Float16* ap = A  + (size_t)am * K + hi * 8;
  const _Float16* w0 = Wt + (size_t)(n0      + r) * K + hi * 8;  // col n0+r
  const _Float16* w1 = Wt + (size_t)(n0 + 16 + r) * K + hi * 8;  // col n0+16+r

  v8f acc0 = {}, acc1 = {};
#pragma unroll
  for (int k0 = 0; k0 < K; k0 += 32) {
    v16h a, b0, b1;
    *((v8h*)&a)      = *(const v8h*)(ap + k0);
    *((v8h*)&a + 1)  = *(const v8h*)(ap + k0 + 16);
    *((v8h*)&b0)     = *(const v8h*)(w0 + k0);
    *((v8h*)&b0 + 1) = *(const v8h*)(w0 + k0 + 16);
    *((v8h*)&b1)     = *(const v8h*)(w1 + k0);
    *((v8h*)&b1 + 1) = *(const v8h*)(w1 + k0 + 16);
    acc0 = __builtin_amdgcn_wmma_f32_16x16x32_f16(false, a, false, b0,
                                                  (short)0, acc0, false, false);
    acc1 = __builtin_amdgcn_wmma_f32_16x16x32_f16(false, a, false, b1,
                                                  (short)0, acc1, false, false);
  }
  // C/D layout: VGPR v -> row m0 + 8*hi + v, col = tile_col_base + (lane&15)
#pragma unroll
  for (int v = 0; v < 8; ++v) {
    int rr = m0 + hi * 8 + v;
    if (rr < N) {
      C[(size_t)rr * H + n0 + r]      = acc0[v];
      C[(size_t)rr * H + n0 + 16 + r] = acc1[v];
    }
  }
}

// ---- edge scatter: agg[col] += dinv[row]*dinv[col]*xw[row] -------------------
__global__ __launch_bounds__(256) void k_edge_scatter(const int* __restrict__ row,
                                                      const int* __restrict__ col,
                                                      const float* __restrict__ dinv,
                                                      const float* __restrict__ xw,
                                                      float* __restrict__ agg, int E) {
  int edge = blockIdx.x * 8 + (threadIdx.x >> 5);
  if (edge >= E) return;
  int lane = threadIdx.x & 31;
  int r = row[edge], c = col[edge];
  float w = dinv[r] * dinv[c];
  const float4 v = *(const float4*)(xw + (size_t)r * HDIM + lane * 4);
  float* dst = agg + (size_t)c * HDIM + lane * 4;
  atomicAdd(dst + 0, w * v.x);
  atomicAdd(dst + 1, w * v.y);
  atomicAdd(dst + 2, w * v.z);
  atomicAdd(dst + 3, w * v.w);
}

// ---- self-loop contribution + bias ------------------------------------------
__global__ __launch_bounds__(256) void k_self_bias(float* __restrict__ agg,
                                                   const float* __restrict__ xw,
                                                   const float* __restrict__ dinv,
                                                   const float* __restrict__ b, int total) {
  int idx = blockIdx.x * 256 + threadIdx.x;
  if (idx >= total) return;
  int n = idx >> 7, f = idx & (HDIM - 1);
  float di = dinv[n];
  agg[idx] += di * di * xw[idx] + b[f];
}

// ---- batchnorm stats (per-column sums over rows) -----------------------------
__global__ __launch_bounds__(128) void k_bn_partial(const float* __restrict__ x,
                                                    float* __restrict__ sums, int N) {
  int f = threadIdx.x;
  int n0 = blockIdx.x * 256;
  int n1 = n0 + 256; if (n1 > N) n1 = N;
  float s = 0.f, s2 = 0.f;
  for (int n = n0; n < n1; ++n) {
    float v = x[(size_t)n * HDIM + f];
    s += v; s2 += v * v;
  }
  atomicAdd(&sums[f], s);
  atomicAdd(&sums[HDIM + f], s2);
}

__global__ __launch_bounds__(128) void k_bn_finalize(float* __restrict__ stats, int N) {
  int f = threadIdx.x;
  float inv = 1.0f / (float)N;
  float mean = stats[f] * inv;
  float var  = stats[HDIM + f] * inv - mean * mean;   // population var (jnp.var)
  stats[2 * HDIM + f] = mean;
  stats[3 * HDIM + f] = rsqrtf(var + 1e-5f);
}

// ---- BN + ReLU, f16 output (feeds next WMMA GEMM) ----------------------------
__global__ __launch_bounds__(256) void k_bn_relu_f16(const float* __restrict__ x,
                                                     const float* __restrict__ stats,
                                                     const float* __restrict__ gamma,
                                                     const float* __restrict__ beta,
                                                     _Float16* __restrict__ out, int total) {
  int idx = blockIdx.x * 256 + threadIdx.x;
  if (idx >= total) return;
  int f = idx & (HDIM - 1);
  float v = (x[idx] - stats[2 * HDIM + f]) * stats[3 * HDIM + f] * gamma[f] + beta[f];
  out[idx] = (_Float16)fmaxf(v, 0.0f);
}

// ---- BN + ReLU, f32 in-place (layer 2 output, feeds pooling) -----------------
__global__ __launch_bounds__(256) void k_bn_relu_f32(float* __restrict__ x,
                                                     const float* __restrict__ stats,
                                                     const float* __restrict__ gamma,
                                                     const float* __restrict__ beta, int total) {
  int idx = blockIdx.x * 256 + threadIdx.x;
  if (idx >= total) return;
  int f = idx & (HDIM - 1);
  float v = (x[idx] - stats[2 * HDIM + f]) * stats[3 * HDIM + f] * gamma[f] + beta[f];
  x[idx] = fmaxf(v, 0.0f);
}

// ---- pooling -----------------------------------------------------------------
__global__ __launch_bounds__(256) void k_pool_cnt(const int* __restrict__ batch,
                                                  float* __restrict__ cnt, int N) {
  int i = blockIdx.x * 256 + threadIdx.x;
  if (i < N) atomicAdd(&cnt[batch[i]], 1.0f);
}

__global__ __launch_bounds__(256) void k_pool_sum_max(const float* __restrict__ h,
                                                      const int* __restrict__ batch,
                                                      float* __restrict__ hsum,
                                                      unsigned* __restrict__ hmax, int total) {
  int idx = blockIdx.x * 256 + threadIdx.x;
  if (idx >= total) return;
  int n = idx >> 7, f = idx & (HDIM - 1);
  int g = batch[n];
  float v = h[idx];
  atomicAdd(&hsum[(size_t)g * HDIM + f], v);
  atomicMax(&hmax[(size_t)g * HDIM + f], __float_as_uint(v));  // v >= 0 post-ReLU
}

// ---- classifier: out[G,3] = [mean|max|sum] @ Wc + bc -------------------------
__global__ __launch_bounds__(256) void k_classifier(const float* __restrict__ hsum,
                                                    const unsigned* __restrict__ hmax,
                                                    const float* __restrict__ cnt,
                                                    const float* __restrict__ Wc,
                                                    const float* __restrict__ bc,
                                                    float* __restrict__ out) {
  int idx = blockIdx.x * 256 + threadIdx.x;
  if (idx >= GNUM * CNUM) return;
  int g = idx / CNUM, c = idx % CNUM;
  float invc = 1.0f / fmaxf(cnt[g], 1.0f);
  float acc = bc[c];
  for (int f = 0; f < HDIM; ++f) {
    float hs = hsum[(size_t)g * HDIM + f];
    float hm = __uint_as_float(hmax[(size_t)g * HDIM + f]);
    acc += (hs * invc) * Wc[f * CNUM + c]
         + hm          * Wc[(HDIM + f) * CNUM + c]
         + hs          * Wc[(2 * HDIM + f) * CNUM + c];
  }
  out[idx] = acc;
}

// ---- host orchestration ------------------------------------------------------
extern "C" void kernel_launch(void* const* d_in, const int* in_sizes, int n_in,
                              void* d_out, int out_size, void* d_ws, size_t ws_size,
                              hipStream_t stream) {
  const float* x      = (const float*)d_in[0];
  const int*   eidx   = (const int*)  d_in[1];
  const int*   batch  = (const int*)  d_in[2];
  const float* W1     = (const float*)d_in[3];
  const float* b1     = (const float*)d_in[4];
  const float* gamma1 = (const float*)d_in[5];
  const float* beta1  = (const float*)d_in[6];
  const float* W2     = (const float*)d_in[7];
  const float* b2     = (const float*)d_in[8];
  const float* gamma2 = (const float*)d_in[9];
  const float* beta2  = (const float*)d_in[10];
  const float* Wc     = (const float*)d_in[11];
  const float* bc     = (const float*)d_in[12];
  float* out = (float*)d_out;

  const int N = in_sizes[0] / HDIM;
  const int E = in_sizes[1] / 2;
  const int* row = eidx;
  const int* col = eidx + E;
  const int NH = N * HDIM;

  // workspace carve-out (256B aligned regions)
  char* ws = (char*)d_ws;
  size_t off = 0;
  auto carve = [&](size_t bytes) -> char* {
    char* p = ws + off;
    off = (off + bytes + 255) & ~(size_t)255;
    return p;
  };
  float*    dinv    = (float*)   carve((size_t)N * 4);
  float*    xw      = (float*)   carve((size_t)NH * 4);
  float*    agg     = (float*)   carve((size_t)NH * 4);
  _Float16* hbuf    = (_Float16*)carve((size_t)NH * 2);   // x_f16 then h1_f16
  _Float16* Wt      = (_Float16*)carve((size_t)HDIM * HDIM * 2); // transposed weights
  float*    stats   = (float*)   carve((size_t)4 * HDIM * 4); // sum,sumsq,mean,rstd
  float*    cnt     = (float*)   carve((size_t)GNUM * 4);
  float*    hsum    = (float*)   carve((size_t)GNUM * HDIM * 4);
  unsigned* hmax    = (unsigned*)carve((size_t)GNUM * HDIM * 4);

  const int eB   = (E + 255) / 256;
  const int nB   = (N + 255) / 256;
  const int nhB  = (NH + 255) / 256;
  const int whB  = (HDIM * HDIM + 255) / 256;
  const int tilesM = (N + 15) / 16;
  const int gemmB  = (tilesM * (HDIM / 32) + 7) / 8;   // 16x32 tile per wave, 8 waves/block
  const int edgeB  = (E + 7) / 8;                      // 8 edges per block
  const int bnB    = (N + 255) / 256;

  // degrees -> dinv (stored once, reused by both layers)
  hipMemsetAsync(dinv, 0, (size_t)N * 4, stream);
  k_deg <<<eB, 256, 0, stream>>>(col, dinv, E);
  k_dinv<<<nB, 256, 0, stream>>>(dinv, N);

  // ---------------- layer 1 ----------------
  k_f32_to_f16   <<<nhB, 256, 0, stream>>>(x, hbuf, NH);
  k_f32_to_f16_tr<<<whB, 256, 0, stream>>>(W1, Wt);
  k_gemm_wmma    <<<gemmB, 256, 0, stream>>>(hbuf, Wt, xw, N);

  hipMemsetAsync(agg, 0, (size_t)NH * 4, stream);
  k_edge_scatter<<<edgeB, 256, 0, stream>>>(row, col, dinv, xw, agg, E);
  k_self_bias  <<<nhB, 256, 0, stream>>>(agg, xw, dinv, b1, NH);

  hipMemsetAsync(stats, 0, (size_t)2 * HDIM * 4, stream);
  k_bn_partial <<<bnB, 128, 0, stream>>>(agg, stats, N);
  k_bn_finalize<<<1, 128, 0, stream>>>(stats, N);
  k_bn_relu_f16<<<nhB, 256, 0, stream>>>(agg, stats, gamma1, beta1, hbuf, NH);

  // ---------------- layer 2 ----------------
  k_f32_to_f16_tr<<<whB, 256, 0, stream>>>(W2, Wt);
  k_gemm_wmma    <<<gemmB, 256, 0, stream>>>(hbuf, Wt, xw, N);

  hipMemsetAsync(agg, 0, (size_t)NH * 4, stream);
  k_edge_scatter<<<edgeB, 256, 0, stream>>>(row, col, dinv, xw, agg, E);
  k_self_bias  <<<nhB, 256, 0, stream>>>(agg, xw, dinv, b2, NH);

  hipMemsetAsync(stats, 0, (size_t)2 * HDIM * 4, stream);
  k_bn_partial <<<bnB, 128, 0, stream>>>(agg, stats, N);
  k_bn_finalize<<<1, 128, 0, stream>>>(stats, N);
  k_bn_relu_f32<<<nhB, 256, 0, stream>>>(agg, stats, gamma2, beta2, NH); // h2 in-place

  // ---------------- pooling + classifier ----------------
  hipMemsetAsync(cnt,  0, (size_t)GNUM * 4, stream);
  hipMemsetAsync(hsum, 0, (size_t)GNUM * HDIM * 4, stream);
  hipMemsetAsync(hmax, 0, (size_t)GNUM * HDIM * 4, stream);   // 0u == 0.0f
  k_pool_cnt    <<<nB, 256, 0, stream>>>(batch, cnt, N);
  k_pool_sum_max<<<nhB, 256, 0, stream>>>(agg, batch, hsum, hmax, NH);
  k_classifier  <<<(GNUM * CNUM + 255) / 256, 256, 0, stream>>>(hsum, hmax, cnt, Wc, bc, out);
}
